// GptOssAttention_43542378447344
// MI455X (gfx1250) — matrix operand order repaired and checked by reading, not compiled, not run on hardware
//
#include <hip/hip_runtime.h>
#include <hip/hip_bf16.h>

// Problem constants
#define S_    4096
#define HID_  2880
#define HKV_  8
#define D_    64
#define W_    128
#define B_    32
#define QKVO_ 5120   // (64+16)*64
#define QD_   4096   // H_Q*D
#define NPAD2 3072   // HID_ rounded up to 256 for GEMM2 B tiles

typedef __attribute__((ext_vector_type(16))) __bf16 bf16x16;
typedef __attribute__((ext_vector_type(8)))  float  f32x8;

union AFrag { bf16x16 v; unsigned int u[8]; };

__device__ __forceinline__ f32x8 zero8() {
  f32x8 z; for (int i = 0; i < 8; ++i) z[i] = 0.0f; return z;
}
__device__ __forceinline__ unsigned short f2bf(float f) {
  union { float f; unsigned int u; } x; x.f = f;
  unsigned int r = x.u + 0x7FFFu + ((x.u >> 16) & 1u);  // round-nearest-even
  return (unsigned short)(r >> 16);
}
__device__ __forceinline__ float bf2f(unsigned short h) {
  union { unsigned int u; float f; } x; x.u = ((unsigned int)h) << 16; return x.f;
}
__device__ __forceinline__ unsigned int pack2(float lo, float hi) {
  return (unsigned int)f2bf(lo) | ((unsigned int)f2bf(hi) << 16);
}

// ---------------- CDNA5 async global->LDS copy path ------------------------
#if __has_builtin(__builtin_amdgcn_global_load_async_to_lds_b128)
#define ASYNC_COPY 1
#endif

typedef int v4i_t __attribute__((vector_size(16)));
typedef __attribute__((address_space(1))) v4i_t* gv4i_p;   // <4 x i32> addrspace(1)*
typedef __attribute__((address_space(3))) v4i_t* lv4i_p;   // <4 x i32> addrspace(3)*

#ifdef ASYNC_COPY
typedef gv4i_p GP;
typedef lv4i_p LP;
template <int OFF>
__device__ __forceinline__ void copyo(GP g, LP l) {
  __builtin_amdgcn_global_load_async_to_lds_b128(g, l, OFF, 0);
}
#if __has_builtin(__builtin_amdgcn_s_wait_asynccnt)
#define WAIT_ASYNC(n) __builtin_amdgcn_s_wait_asynccnt(n)
#else
#define WAIT_ASYNC(n) asm volatile("s_wait_asynccnt %0" ::"n"(n) : "memory")
#endif
#else
typedef const uint4* GP;
typedef uint4* LP;
template <int OFF>
__device__ __forceinline__ void copyo(GP g, LP l) {
  *(uint4*)((char*)l + OFF) = *(const uint4*)((const char*)g + OFF);
}
#define WAIT_ASYNC(n) do {} while (0)
#endif

// ---------------------------------------------------------------------------
// fp32 -> bf16 (same layout), vectorized float4 -> 2x packed u32
// ---------------------------------------------------------------------------
__global__ __launch_bounds__(256)
void convert_kernel(const float* __restrict__ src, unsigned short* __restrict__ dst,
                    int n4) {
  int i = blockIdx.x * 256 + threadIdx.x;
  if (i < n4) {
    float4 v = ((const float4*)src)[i];
    uint2 o; o.x = pack2(v.x, v.y); o.y = pack2(v.z, v.w);
    ((uint2*)dst)[i] = o;
  }
}

// ---------------------------------------------------------------------------
// fp32 [K][N] -> bf16 [Npad][K] (transpose+convert), zero pad for n >= N
// ---------------------------------------------------------------------------
__global__ __launch_bounds__(256)
void transpose_convert_kernel(const float* __restrict__ src,
                              unsigned short* __restrict__ dst,
                              int K, int N) {
  __shared__ float T[32][33];
  const int n0 = blockIdx.x * 32;
  const int k0 = blockIdx.y * 32;
  const int tid = threadIdx.x;
  for (int i = 0; i < 4; ++i) {
    int e = tid + i * 256;            // e = kk*32 + nn
    int kk = e >> 5, nn = e & 31;
    int n = n0 + nn;
    T[nn][kk] = (n < N) ? src[(size_t)(k0 + kk) * N + n] : 0.0f;
  }
  __syncthreads();
  for (int i = 0; i < 4; ++i) {
    int e = tid + i * 256;            // e = nn*32 + kk
    int nn = e >> 5, kk = e & 31;
    dst[(size_t)(n0 + nn) * K + (k0 + kk)] = f2bf(T[nn][kk]);
  }
}

// ---------------------------------------------------------------------------
// C = A(MxK,bf16) @ Bt(N x K,bf16 pre-transposed) + bias.
// Block tile 128x256x32, double-buffered LDS fed by async global->LDS copies.
// 8 waves, each computing 64x64 (4x4 WMMA tiles = 16 WMMA / K-step).
// Templated output (f32 or bf16) to keep the epilogue branch-free.
// ---------------------------------------------------------------------------
#define TM 128
#define TN 256
#define TK 32

template <bool BF16OUT>
__global__ __launch_bounds__(256, 1)
void gemm_bf16_kernel(const unsigned short* __restrict__ A,
                      const unsigned short* __restrict__ Bt,
                      const float* __restrict__ bias,
                      void* __restrict__ Cout,
                      int M, int N, int K) {
  __shared__ unsigned short As[2][TM * TK];   // 16 KB
  __shared__ unsigned short Bs[2][TN * TK];   // 32 KB
  const int tid  = threadIdx.x;
  const int lane = tid & 31;
  const int wave = tid >> 5;
  const int wm   = wave >> 2;     // 0..1 (64-row group)
  const int wn   = wave & 3;      // 0..3 (64-col group)
  const int l15  = lane & 15;
  const int kg   = lane >> 4;
  const int bm   = blockIdx.y * TM;
  const int bn   = blockIdx.x * TN;

  f32x8 acc[4][4];
  for (int i = 0; i < 4; ++i)
    for (int j = 0; j < 4; ++j) acc[i][j] = zero8();

  // copy assignment (per stage = 24KB):
  //  A: thread covers 32B of row tid>>1 (2 async b128)
  //  B: thread covers full 64B of row tid (4 async b128)
  const int arow = tid >> 1;
  const int acs  = (tid & 1) * 16;           // ushort offset within A row
  GP gA = (GP)(void*)(A  + (size_t)(bm + arow) * K + acs);
  GP gB = (GP)(void*)(Bt + (size_t)(bn + tid) * K);
  LP lA[2] = { (LP)(void*)&As[0][arow * TK + acs], (LP)(void*)&As[1][arow * TK + acs] };
  LP lB[2] = { (LP)(void*)&Bs[0][tid * TK],        (LP)(void*)&Bs[1][tid * TK] };

#define STAGE(kt, buf)                                  \
  do {                                                  \
    GP _ga = gA + (kt) * 4;  /* 4 x 16B = 32 ushorts */ \
    GP _gb = gB + (kt) * 4;                             \
    copyo<0>(_ga, lA[buf]);   copyo<16>(_ga, lA[buf]);  \
    copyo<0>(_gb, lB[buf]);   copyo<16>(_gb, lB[buf]);  \
    copyo<32>(_gb, lB[buf]);  copyo<48>(_gb, lB[buf]);  \
  } while (0)

  auto compute = [&](int cur) {
    AFrag af[4], bf[4];
    for (int j = 0; j < 4; ++j) {
      int col = wn * 64 + j * 16 + l15;
      const unsigned short* p = &Bs[cur][col * TK + kg * 16];
      for (int v = 0; v < 8; ++v) bf[j].u[v] = *(const unsigned int*)&p[2 * v];
    }
    for (int i = 0; i < 4; ++i) {
      int row = wm * 64 + i * 16 + l15;
      const unsigned short* p = &As[cur][row * TK];
      for (int v = 0; v < 8; ++v) {
        int kb = kg * 16 + v * 2;
        af[i].u[v] = *(const unsigned int*)&p[kb];
      }
    }
    for (int i = 0; i < 4; ++i)
      for (int j = 0; j < 4; ++j)
        acc[i][j] = __builtin_amdgcn_wmma_f32_16x16x32_bf16(
            false, af[i].v, false, bf[j].v, (short)0, acc[i][j], false, false);
  };

  const int nk = K / TK;
  STAGE(0, 0);
  int kt = 0;
  for (; kt < nk - 1; ++kt) {              // steady state: unconditional body
    STAGE(kt + 1, (kt & 1) ^ 1);
    WAIT_ASYNC(6);                          // stage kt done; kt+1 (6 ops) in flight
    __syncthreads();
    compute(kt & 1);
    __syncthreads();
  }
  WAIT_ASYNC(0);                            // last stage
  __syncthreads();
  compute(kt & 1);
#undef STAGE

  // Branch-free strided epilogue
  for (int i = 0; i < 4; ++i)
    for (int j = 0; j < 4; ++j) {
      int colb = bn + wn * 64 + j * 16 + l15;
      if (colb < N) {
        float bb = bias[colb];
        size_t base = (size_t)(bm + wm * 64 + i * 16 + kg * 8) * N + colb;
        for (int r = 0; r < 8; ++r) {
          float v = acc[i][j][r] + bb;
          if (BF16OUT) ((unsigned short*)Cout)[base + (size_t)r * N] = f2bf(v);
          else         ((float*)Cout)[base + (size_t)r * N] = v;
        }
      }
    }
}

// ---------------------------------------------------------------------------
// NeoX RoPE in place on bf16 qkv (q,k cols [0,4608)); thread per (s,h,i)
// ---------------------------------------------------------------------------
__global__ __launch_bounds__(256)
void rope_kernel(unsigned short* __restrict__ qkv, const float* __restrict__ cosp,
                 const float* __restrict__ sinp) {
  int idx = blockIdx.x * 256 + threadIdx.x;   // exactly 4096*72*32 threads
  int i = idx & 31;
  int h = (idx >> 5) % 72;
  int s = idx / (72 * 32);
  unsigned short* p = qkv + (size_t)s * QKVO_ + h * 64;
  float x1 = bf2f(p[i]), x2 = bf2f(p[i + 32]);
  float c = cosp[s * 32 + i], sn = sinp[s * 32 + i];
  p[i]      = f2bf(x1 * c - x2 * sn);
  p[i + 32] = f2bf(x2 * c + x1 * sn);
}

// ---------------------------------------------------------------------------
// Sliding-window attention with sinks, qkv in bf16, ctx out in bf16.
// grid 2048 = B*HKV*G; 256 threads (8 waves x 16 query rows)
// dyn LDS: Vt[64][256] (32KB) + P[8][16][256] (64KB) = 96KB bf16
// ---------------------------------------------------------------------------
__global__ __launch_bounds__(256)
void attn_kernel(const unsigned short* __restrict__ qkv,
                 const float* __restrict__ sinks,
                 unsigned short* __restrict__ attnb) {
  extern __shared__ unsigned short smem[];
  unsigned short* Vt = smem;              // Vt[d][m]
  unsigned short* P  = smem + 64 * 256;

  const int bidx = blockIdx.x;
  const int b  = bidx >> 6;
  const int kh = (bidx >> 3) & 7;
  const int g  = bidx & 7;
  const int tid  = threadIdx.x;
  const int lane = tid & 31;
  const int wave = tid >> 5;
  const int l15  = lane & 15;
  const int kg   = lane >> 4;

  const int qcol = (kh * 8 + g) * 64;
  const int kcol = QD_ + kh * 64;
  const int vcol = QD_ + HKV_ * 64 + kh * 64;
  const float sk = sinks[kh * 8 + g];
  const float scale = 0.125f;

  // stage V transposed (bf16 passthrough, no conversion)
  {
    int m = tid;
    bool valid = (b > 0) || (m >= 128);
    const unsigned short* vp = qkv + (size_t)(b * 128 + m - 128) * QKVO_ + vcol;
    for (int d = 0; d < 64; ++d)
      Vt[d * 256 + m] = valid ? vp[d] : (unsigned short)0;
  }

  // Q fragments: direct u32 reads of bf16 pairs from global
  const int qbase = wave * 16;
  AFrag qf[2];
  {
    int srow = b * 128 + qbase + l15;
    const unsigned short* qp = qkv + (size_t)srow * QKVO_ + qcol;
    for (int ks = 0; ks < 2; ++ks)
      for (int v = 0; v < 8; ++v) {
        int kb = kg * 16 + v * 2;
        qf[ks].u[v] = *(const unsigned int*)&qp[ks * 32 + kb];
      }
  }

  // scores: 16 key tiles x (2 k-steps over D=64), f32 accum in registers
  f32x8 sacc[16];
  for (int t = 0; t < 16; ++t) {
    f32x8 c = zero8();
    bool valid = (b > 0) || (t >= 8);     // uniform
    if (valid) {
      int m = t * 16 + l15;
      const unsigned short* kp = qkv + (size_t)(b * 128 + m - 128) * QKVO_ + kcol;
      for (int ks = 0; ks < 2; ++ks) {
        AFrag bf;
        for (int v = 0; v < 8; ++v)
          bf.u[v] = *(const unsigned int*)&kp[ks * 32 + kg * 16 + 2 * v];
        c = __builtin_amdgcn_wmma_f32_16x16x32_bf16(
            false, qf[ks].v, false, bf.v, (short)0, c, false, false);
      }
    }
    sacc[t] = c;
  }

  // masked softmax + sink; C-layout row = qbase + r + 8*laneHalf
  float rden[8];
  for (int r = 0; r < 8; ++r) {
    int qi = qbase + r + kg * 8;
    float mx = -3.0e38f;
    float sv[16];
    for (int t = 0; t < 16; ++t) {
      int ki = t * 16 + l15;
      bool ok = (ki >= qi + 1) && (ki <= qi + W_) && ((b > 0) || (ki >= W_));
      float s = sacc[t][r] * scale;
      sv[t] = ok ? s : -3.0e38f;
      mx = fmaxf(mx, sv[t]);
    }
    for (int off = 1; off < 16; off <<= 1) mx = fmaxf(mx, __shfl_xor(mx, off, 32));
    float m = fmaxf(mx, sk);
    float sum = 0.0f;
    unsigned short* Pw = P + ((wave * 16 + r + kg * 8) * 256);
    for (int t = 0; t < 16; ++t) {
      float e = (sv[t] > -1.0e38f) ? __expf(sv[t] - m) : 0.0f;
      sum += e;
      Pw[t * 16 + l15] = f2bf(e);
    }
    for (int off = 1; off < 16; off <<= 1) sum += __shfl_xor(sum, off, 32);
    rden[r] = 1.0f / (sum + __expf(sk - m));
  }
  __syncthreads();

  // ctx = P(16x256) @ V(256x64)
  f32x8 cacc[4];
  for (int j = 0; j < 4; ++j) cacc[j] = zero8();
  const unsigned short* Pw = P + wave * 16 * 256;
  for (int ks = 0; ks < 8; ++ks) {
    AFrag af;
    {
      const unsigned short* p = &Pw[l15 * 256 + ks * 32];
      for (int v = 0; v < 8; ++v) {
        int kb = kg * 16 + v * 2;
        af.u[v] = *(const unsigned int*)&p[kb];
      }
    }
    for (int j = 0; j < 4; ++j) {
      AFrag bf;
      const unsigned short* p = &Vt[(j * 16 + l15) * 256 + ks * 32 + kg * 16];
      for (int v = 0; v < 8; ++v) bf.u[v] = *(const unsigned int*)&p[2 * v];
      cacc[j] = __builtin_amdgcn_wmma_f32_16x16x32_bf16(
          false, af.v, false, bf.v, (short)0, cacc[j], false, false);
    }
  }

  for (int j = 0; j < 4; ++j) {
    int col = qcol + j * 16 + l15;
    for (int r = 0; r < 8; ++r) {
      int srow = b * 128 + qbase + r + kg * 8;
      attnb[(size_t)srow * QD_ + col] = f2bf(cacc[j][r] * rden[r]);
    }
  }
}

// ---------------------------------------------------------------------------
extern "C" void kernel_launch(void* const* d_in, const int* in_sizes, int n_in,
                              void* d_out, int out_size, void* d_ws, size_t ws_size,
                              hipStream_t stream) {
  const float* hs    = (const float*)d_in[0];
  const float* cosp  = (const float*)d_in[1];
  const float* sinp  = (const float*)d_in[2];
  const float* Wqkv  = (const float*)d_in[3];
  const float* bqkv  = (const float*)d_in[4];
  const float* Wo    = (const float*)d_in[5];
  const float* bo    = (const float*)d_in[6];
  const float* sinks = (const float*)d_in[7];
  float* out = (float*)d_out;

  // workspace (~154 MB): qkv bf16, hs bf16, Wqkv^T bf16, Wo^T bf16 (padded), ctx bf16
  char* w = (char*)d_ws;
  unsigned short* qkv   = (unsigned short*)w; w += (size_t)S_ * QKVO_ * 2;
  unsigned short* hsb   = (unsigned short*)w; w += (size_t)S_ * HID_ * 2;
  unsigned short* wqkvt = (unsigned short*)w; w += (size_t)QKVO_ * HID_ * 2;
  unsigned short* wot   = (unsigned short*)w; w += (size_t)NPAD2 * QD_ * 2;
  unsigned short* attnb = (unsigned short*)w; w += (size_t)S_ * QD_ * 2;

  (void)hipFuncSetAttribute((const void*)attn_kernel,
                            hipFuncAttributeMaxDynamicSharedMemorySize, 98304);

  // 0) precondition operands: bf16 convert / transpose-convert
  convert_kernel<<<(S_ * HID_ / 4 + 255) / 256, 256, 0, stream>>>(hs, hsb, S_ * HID_ / 4);
  transpose_convert_kernel<<<dim3(QKVO_ / 32, HID_ / 32), 256, 0, stream>>>(
      Wqkv, wqkvt, HID_, QKVO_);
  transpose_convert_kernel<<<dim3(NPAD2 / 32, QD_ / 32), 256, 0, stream>>>(
      Wo, wot, QD_, HID_);

  // 1) qkv = hs @ Wqkv + b_qkv  (bf16 out)
  gemm_bf16_kernel<true><<<dim3(QKVO_ / TN, S_ / TM), 256, 0, stream>>>(
      hsb, wqkvt, bqkv, qkv, S_, QKVO_, HID_);
  // 2) RoPE in place on q,k
  rope_kernel<<<(S_ * 72 * 32) / 256, 256, 0, stream>>>(qkv, cosp, sinp);
  // 3) attention -> ctx bf16
  attn_kernel<<<B_ * HKV_ * 8, 256, 98304, stream>>>(qkv, sinks, attnb);
  // 4) out = ctx @ Wo + b_o (f32 out)
  gemm_bf16_kernel<false><<<dim3((HID_ + TN - 1) / TN, S_ / TM), 256, 0, stream>>>(
      attnb, wot, bo, out, S_, HID_, QD_);
}
